// FlashAttentionWithKVCache_90297392431790
// MI455X (gfx1250) — matrix operand order repaired and checked by reading, not compiled, not run on hardware
//
#include <hip/hip_runtime.h>
#include <hip/hip_bf16.h>
#include <math.h>

typedef __attribute__((ext_vector_type(16))) __bf16 v16bf;
typedef __attribute__((ext_vector_type(8)))  __bf16 v8bf;
typedef __attribute__((ext_vector_type(4)))  __bf16 v4bf;
typedef __attribute__((ext_vector_type(8)))  float  v8f;

#define WMMA_BF16(a, b, c) \
    __builtin_amdgcn_wmma_f32_16x16x32_bf16(false, (a), false, (b), (short)0, (c), false, false)

static constexpr int BATCH  = 2;
static constexpr int SEQ    = 2048;
static constexpr int DMODEL = 1024;
static constexpr int NHEAD  = 16;
static constexpr int HDIM   = 64;
static constexpr float NEG_INF = -1000000000.0f;

// ---------------------------------------------------------------------------
// bf16 fragment loaders: every fragment is two 16B vector loads.
// Layouts per CDNA5 ISA (16-bit A 16x32, 16-bit B 32x16).
// ---------------------------------------------------------------------------
__device__ inline v16bf cat8(v8bf lo, v8bf hi) {
    return __builtin_shufflevector(lo, hi, 0, 1, 2, 3, 4, 5, 6, 7,
                                           8, 9, 10, 11, 12, 13, 14, 15);
}

// A fragment 16x32 (MxK) from row-major bf16 A[m][k], row stride lda.
// Lane l: row m (passed in); idx 0..7 -> K=kb+half*8+idx, idx 8..15 -> +16.
__device__ inline v16bf load_a_bf(const __bf16* __restrict__ A, int lda, int m, int kb, int lane) {
    const int half = lane >> 4;
    const __bf16* p = A + (size_t)m * lda + kb;
    v8bf lo = *reinterpret_cast<const v8bf*>(p + half * 8);
    v8bf hi = *reinterpret_cast<const v8bf*>(p + 16 + half * 8);
    return cat8(lo, hi);
}

// B fragment 32x16 (KxN) where memory holds M[N][K] (K contiguous), row stride ldm.
// Lane l: n = nbase + (l&15); idx 0..7 -> K = kb + (l>>4)*8 + idx, idx 8..15 -> +16.
__device__ inline v16bf load_bT_bf(const __bf16* __restrict__ M, int ldm, int kb, int nbase, int lane) {
    const int half = lane >> 4;
    const __bf16* p = M + (size_t)(nbase + (lane & 15)) * ldm + kb;
    v8bf lo = *reinterpret_cast<const v8bf*>(p + half * 8);
    v8bf hi = *reinterpret_cast<const v8bf*>(p + 16 + half * 8);
    return cat8(lo, hi);
}

// ---------------------------------------------------------------------------
// Prep kernels (one-time): fp32 -> bf16 convert, and transpose-convert of
// weights into [N][K] bf16 so GEMM B-fragments are K-contiguous.
// ---------------------------------------------------------------------------
__global__ __launch_bounds__(256) void cvt_bf16_kernel(
    const float* __restrict__ in, __bf16* __restrict__ out, int n4) {
    int i = blockIdx.x * blockDim.x + threadIdx.x;
    if (i < n4) {
        float4 f = reinterpret_cast<const float4*>(in)[i];
        v4bf o;
        o[0] = (__bf16)f.x; o[1] = (__bf16)f.y; o[2] = (__bf16)f.z; o[3] = (__bf16)f.w;
        reinterpret_cast<v4bf*>(out)[i] = o;
    }
}

// W: K x N fp32 row-major  ->  Wt: N x K bf16 row-major. 32x32 LDS tiles.
__global__ __launch_bounds__(256) void transpose_cvt_kernel(
    const float* __restrict__ W, __bf16* __restrict__ Wt, int K, int N) {
    __shared__ float tile[32][33];
    const int tx = threadIdx.x & 31;
    const int ty = threadIdx.x >> 5;             // 0..7
    const int nt = blockIdx.x * 32;
    const int kt = blockIdx.y * 32;
#pragma unroll
    for (int i = ty; i < 32; i += 8)
        tile[i][tx] = W[(size_t)(kt + i) * N + nt + tx];
    __syncthreads();
#pragma unroll
    for (int i = ty; i < 32; i += 8)
        Wt[(size_t)(nt + i) * K + kt + tx] = (__bf16)tile[tx][i];
}

// ---------------------------------------------------------------------------
// Kernel 1: QKV = x @ W_attn with fused RoPE epilogue.
//   q -> bf16 (B,H,S,64), pre-scaled by 1/sqrt(HD)
//   k -> bf16 (B,H,S,64), roped
//   v -> bf16 TRANSPOSED (B,H,64,S) so P@V B-fragments are K-contiguous.
// One wave per 16x64 output strip; all fragment loads are b128.
// ---------------------------------------------------------------------------
__global__ __launch_bounds__(128) void qkv_rope_kernel(
    const __bf16* __restrict__ xb, const __bf16* __restrict__ Wt,
    __bf16* __restrict__ qout, __bf16* __restrict__ kout, __bf16* __restrict__ vtout) {
    const int lane = threadIdx.x & 31;
    const int wave = blockIdx.x * (blockDim.x >> 5) + (threadIdx.x >> 5);
    const int NW = (3 * DMODEL) / 64;   // 48 wave-columns
    const int mt = wave / NW;
    const int nw = wave % NW;
    const int mbase = mt * 16;
    const int nbase = nw * 64;

    v8f zero = {};
    v8f acc[4] = {zero, zero, zero, zero};

    const int m = mbase + (lane & 15);
    for (int kb = 0; kb < DMODEL; kb += 32) {
        v16bf a = load_a_bf(xb, DMODEL, m, kb, lane);
#pragma unroll
        for (int t = 0; t < 4; ++t) {
            v16bf b = load_bT_bf(Wt, DMODEL, kb, nbase + t * 16, lane);
            acc[t] = WMMA_BF16(a, b, acc[t]);
        }
    }

    const int half = lane >> 4;
    const int sec  = nbase / DMODEL;             // 0=q, 1=k, 2=v (uniform per wave)
    const int h    = (nbase % DMODEL) / HDIM;    // head (uniform per wave)

#pragma unroll
    for (int t = 0; t < 4; ++t) {
        const int dim = t * 16 + (lane & 15);
#pragma unroll
        for (int r = 0; r < 8; ++r) {
            const int row = mbase + r + 8 * half;
            const int b_  = row / SEQ;
            const int s_  = row % SEQ;
            const int bh  = b_ * NHEAD + h;
            float val = acc[t][r];
            if (sec == 2) {
                // v: transposed store (B,H,dim,s)
                vtout[((size_t)bh * HDIM + dim) * SEQ + s_] = (__bf16)val;
            } else {
                // RoPE: out[2i] = x[2i]*cos + x[2i+1]*sin ; out[2i+1] = x[2i+1]*cos - x[2i]*sin
                float partner = __shfl_xor(val, 1, 32);
                float theta = (float)s_ * __powf(10000.0f, -(float)(dim & ~1) / 64.0f);
                float sn, cs;
                __sincosf(theta, &sn, &cs);
                float outv = (dim & 1) ? (val * cs - partner * sn) : (val * cs + partner * sn);
                if (sec == 0) outv *= 0.125f;    // 1/sqrt(HD)
                __bf16* dst = (sec == 0) ? qout : kout;
                dst[((size_t)bh * SEQ + s_) * HDIM + dim] = (__bf16)outv;
            }
        }
    }
}

// ---------------------------------------------------------------------------
// Kernel 2: causal flash attention, one wave (32-thread block) per
// (b, h, 16-query tile); keys streamed in chunks of 32 with online softmax.
// All global fragment loads are b128 bf16; P bounces through LDS as bf16.
// ---------------------------------------------------------------------------
__global__ __launch_bounds__(32) void attn_kernel(
    const __bf16* __restrict__ qa, const __bf16* __restrict__ ka,
    const __bf16* __restrict__ vt, __bf16* __restrict__ y) {
    __shared__ __align__(16) __bf16 plds[16 * 32];

    const int lane  = threadIdx.x & 31;
    const int half  = lane >> 4;
    const int qt    = blockIdx.x % (SEQ / 16);
    const int bh    = blockIdx.x / (SEQ / 16);
    const int qbase = qt * 16;

    const __bf16* qp  = qa + (size_t)bh * SEQ * HDIM;
    const __bf16* kp  = ka + (size_t)bh * SEQ * HDIM;
    const __bf16* vtp = vt + (size_t)bh * HDIM * SEQ;   // (64, S) per head

    // q fragments (constant across key chunks): head-dim chunks [0,32), [32,64)
    const v16bf aq0 = load_a_bf(qp, HDIM, qbase + (lane & 15), 0,  lane);
    const v16bf aq1 = load_a_bf(qp, HDIM, qbase + (lane & 15), 32, lane);

    v8f zero = {};
    v8f acc[4] = {zero, zero, zero, zero};
    float mrun[8], lrun[8];
#pragma unroll
    for (int r = 0; r < 8; ++r) { mrun[r] = -1.0e30f; lrun[r] = 0.0f; }

    const int nchunks = (qbase + 16 + 31) >> 5;   // causal truncation
    for (int kc = 0; kc < nchunks; ++kc) {
        const int kb0 = kc * 32;

        // scores: q(16x64) @ k^T(64x32) -> two 16x16 f32 tiles
        v8f s0 = zero, s1 = zero;
        {
            v16bf b00 = load_bT_bf(kp, HDIM, 0,  kb0,      lane);
            v16bf b01 = load_bT_bf(kp, HDIM, 32, kb0,      lane);
            v16bf b10 = load_bT_bf(kp, HDIM, 0,  kb0 + 16, lane);
            v16bf b11 = load_bT_bf(kp, HDIM, 32, kb0 + 16, lane);
            s0 = WMMA_BF16(aq0, b00, s0);
            s0 = WMMA_BF16(aq1, b01, s0);
            s1 = WMMA_BF16(aq0, b10, s1);
            s1 = WMMA_BF16(aq1, b11, s1);
        }

        const int key0 = kb0 + (lane & 15);
        const int key1 = key0 + 16;

        // mask + online softmax; P written to LDS (16x32 row-major bf16)
#pragma unroll
        for (int r = 0; r < 8; ++r) {
            const int qidx = qbase + r + 8 * half;
            float v0 = s0[r] + ((key0 > qidx) ? NEG_INF : 0.0f);
            float v1 = s1[r] + ((key1 > qidx) ? NEG_INF : 0.0f);

            float mx = fmaxf(v0, v1);
            mx = fmaxf(mx, __shfl_xor(mx, 1, 32));
            mx = fmaxf(mx, __shfl_xor(mx, 2, 32));
            mx = fmaxf(mx, __shfl_xor(mx, 4, 32));
            mx = fmaxf(mx, __shfl_xor(mx, 8, 32));

            const float mnew = fmaxf(mrun[r], mx);
            const float corr = __expf(mrun[r] - mnew);
            const float p0 = __expf(v0 - mnew);
            const float p1 = __expf(v1 - mnew);

            float ps = p0 + p1;
            ps += __shfl_xor(ps, 1, 32);
            ps += __shfl_xor(ps, 2, 32);
            ps += __shfl_xor(ps, 4, 32);
            ps += __shfl_xor(ps, 8, 32);

            lrun[r] = lrun[r] * corr + ps;
            mrun[r] = mnew;
#pragma unroll
            for (int t = 0; t < 4; ++t) acc[t][r] *= corr;

            const int prow = r + 8 * half;
            plds[prow * 32 + (lane & 15)]      = (__bf16)p0;
            plds[prow * 32 + 16 + (lane & 15)] = (__bf16)p1;
        }
        __syncthreads();   // single-wave workgroup: near-free; orders LDS w->r

        // reload P as A-fragment (16 queries x 32 keys) via ds b128 loads
        v16bf pa;
        {
            const __bf16* prow = plds + (lane & 15) * 32;
            v8bf lo = *reinterpret_cast<const v8bf*>(prow + half * 8);
            v8bf hi = *reinterpret_cast<const v8bf*>(prow + 16 + half * 8);
            pa = cat8(lo, hi);
        }
        __syncthreads();   // protect LDS from next iteration's writes

        // acc += P(16x32) @ V(32x64); V is transposed so K (keys) is contiguous
#pragma unroll
        for (int t = 0; t < 4; ++t) {
            v16bf bv = load_bT_bf(vtp, SEQ, kb0, t * 16, lane);
            acc[t] = WMMA_BF16(pa, bv, acc[t]);
        }
    }

    // y in (B, S, H*HD) bf16 for the projection GEMM
    const int b_ = bh / NHEAD;
    const int h_ = bh % NHEAD;
#pragma unroll
    for (int t = 0; t < 4; ++t) {
        const int dim = t * 16 + (lane & 15);
#pragma unroll
        for (int r = 0; r < 8; ++r) {
            const int s_ = qbase + r + 8 * half;
            y[((size_t)b_ * SEQ + s_) * DMODEL + h_ * HDIM + dim] = (__bf16)(acc[t][r] / lrun[r]);
        }
    }
}

// ---------------------------------------------------------------------------
// Kernel 3: out = y @ W_proj (bf16 in, fp32 out). One wave per 16x64 strip.
// ---------------------------------------------------------------------------
__global__ __launch_bounds__(128) void proj_kernel(
    const __bf16* __restrict__ A, const __bf16* __restrict__ Wt, float* __restrict__ C) {
    const int lane = threadIdx.x & 31;
    const int wave = blockIdx.x * (blockDim.x >> 5) + (threadIdx.x >> 5);
    const int NW = DMODEL / 64;   // 16 wave-columns
    const int mt = wave / NW;
    const int nw = wave % NW;
    const int mbase = mt * 16;
    const int nbase = nw * 64;

    v8f zero = {};
    v8f acc[4] = {zero, zero, zero, zero};

    const int m = mbase + (lane & 15);
    for (int kb = 0; kb < DMODEL; kb += 32) {
        v16bf a = load_a_bf(A, DMODEL, m, kb, lane);
#pragma unroll
        for (int t = 0; t < 4; ++t) {
            v16bf b = load_bT_bf(Wt, DMODEL, kb, nbase + t * 16, lane);
            acc[t] = WMMA_BF16(a, b, acc[t]);
        }
    }

    const int half = lane >> 4;
#pragma unroll
    for (int t = 0; t < 4; ++t) {
#pragma unroll
        for (int r = 0; r < 8; ++r) {
            C[(size_t)(mbase + r + 8 * half) * DMODEL + nbase + t * 16 + (lane & 15)] = acc[t][r];
        }
    }
}

// ---------------------------------------------------------------------------

extern "C" void kernel_launch(void* const* d_in, const int* in_sizes, int n_in,
                              void* d_out, int out_size, void* d_ws, size_t ws_size,
                              hipStream_t stream) {
    (void)in_sizes; (void)n_in; (void)out_size; (void)ws_size;
    const float* x      = (const float*)d_in[0];
    const float* W_attn = (const float*)d_in[1];
    const float* W_proj = (const float*)d_in[2];
    // d_in[3]/d_in[4] (cache_k/cache_v) are all-zero and fully overwritten by
    // this S==cache_len prefill: k_full == roped k, v_full == v. They are
    // inputs (must not be mutated) — roped k / v live in workspace instead.

    __bf16* ws = (__bf16*)d_ws;
    const size_t nx   = (size_t)BATCH * SEQ * DMODEL;   // 4M
    const size_t nwa  = (size_t)3 * DMODEL * DMODEL;    // 3M
    const size_t nwp  = (size_t)DMODEL * DMODEL;        // 1M
    const size_t nper = (size_t)BATCH * NHEAD * SEQ * HDIM;  // 4M

    __bf16* xb  = ws;                 // x as bf16, (B*S, D)
    __bf16* Wta = xb  + nx;           // W_attn^T bf16, (3D, D)
    __bf16* Wtp = Wta + nwa;          // W_proj^T bf16, (D, D)
    __bf16* qb  = Wtp + nwp;          // (B,H,S,64)
    __bf16* kbf = qb  + nper;         // (B,H,S,64)
    __bf16* vtb = kbf + nper;         // (B,H,64,S)  transposed
    __bf16* yb  = vtb + nper;         // (B*S, D)

    // --- prep: bf16 conversion + weight transposes (one-time, ~30 MB traffic)
    cvt_bf16_kernel<<<(int)(nx / 4 + 255) / 256, 256, 0, stream>>>(x, xb, (int)(nx / 4));
    {
        dim3 g(3 * DMODEL / 32, DMODEL / 32);
        transpose_cvt_kernel<<<g, 256, 0, stream>>>(W_attn, Wta, DMODEL, 3 * DMODEL);
    }
    {
        dim3 g(DMODEL / 32, DMODEL / 32);
        transpose_cvt_kernel<<<g, 256, 0, stream>>>(W_proj, Wtp, DMODEL, DMODEL);
    }

    // --- QKV + RoPE: 256 row-tiles x 48 wave-columns = 12288 waves
    {
        const int waves = ((BATCH * SEQ) / 16) * ((3 * DMODEL) / 64);
        qkv_rope_kernel<<<waves / 4, 128, 0, stream>>>(xb, Wta, qb, kbf, vtb);
    }

    // --- attention: one wave per (b, h, 16-query tile)
    attn_kernel<<<BATCH * NHEAD * (SEQ / 16), 32, 0, stream>>>(qb, kbf, vtb, yb);

    // --- output projection: 256 x 16 = 4096 waves
    {
        const int waves = ((BATCH * SEQ) / 16) * (DMODEL / 64);
        proj_kernel<<<waves / 4, 128, 0, stream>>>(yb, Wtp, (float*)d_out);
    }
}
